// EnhancedIrisNet_36077725286552
// MI455X (gfx1250) — matrix operand order repaired and checked by reading, not compiled, not run on hardware
//
#include <hip/hip_runtime.h>
#include <hip/hip_bf16.h>

typedef __attribute__((ext_vector_type(16))) _Float16 v16h;
typedef __attribute__((ext_vector_type(8)))  float    v8f;
typedef __attribute__((ext_vector_type(4)))  unsigned int tdm_v4u;
typedef __attribute__((ext_vector_type(8)))  int          tdm_v8i;
typedef __attribute__((ext_vector_type(4)))  int          tdm_v4i;

// Problem sizes
constexpr int B_   = 64;
constexpr int C_   = 10;
constexpr int HW_  = 900;    // 30*30
constexpr int Lq   = 900;
constexpr int Dm   = 64;
constexpr int NH_  = 4;
constexpr int HD_  = 16;
constexpr int QT_  = 57;     // ceil(900/16)
constexpr int SCOL = 928;    // padded key width: 29 * 32
constexpr int CWPT = 57;     // ceil(16*900 / 256) color-weight elements per thread

// Output layout (floats)
constexpr size_t OUT_PRED = 0;                       // [64,10,30,30] = 576000
constexpr size_t OUT_CMAP = 576000;                  // [64,10,10]    = 6400
constexpr size_t OUT_CW   = 582400;                  // [64,900,900]  = 51840000
constexpr size_t OUT_MAP  = 52422400;                // [64,10,30,30] = 576000

static __device__ inline v8f wmma_f16(v16h a, v16h b, v8f c) {
  return __builtin_amdgcn_wmma_f32_16x16x32_f16(false, a, false, b, (short)0, c, false, false);
}

// ---------------- argmax over channels ----------------
__global__ void k_argmax(const float* __restrict__ in, int* __restrict__ idx) {
  int t = blockIdx.x * blockDim.x + threadIdx.x;
  if (t >= B_ * HW_) return;
  int b = t / HW_, s = t % HW_;
  const float* p = in + (size_t)b * C_ * HW_ + s;
  float best = p[0]; int bi = 0;
  #pragma unroll
  for (int c = 1; c < C_; ++c) {
    float v = p[(size_t)c * HW_];
    if (v > best) { best = v; bi = c; }
  }
  idx[t] = bi;
}

// ---------------- embed gather -> f16 ----------------
__global__ void k_gather(const float* __restrict__ embed, const int* __restrict__ idx,
                         _Float16* __restrict__ xh) {
  int t = blockIdx.x * blockDim.x + threadIdx.x;
  if (t >= B_ * HW_ * Dm) return;
  int d = t % Dm; int bl = t / Dm;
  xh[t] = (_Float16)embed[idx[bl] * Dm + d];
}

// ---------------- weight f32->f16 conversion ----------------
__global__ void k_cvt(const float* __restrict__ ipw, const float* __restrict__ opw,
                      _Float16* __restrict__ ipw16, _Float16* __restrict__ opw16) {
  int t = blockIdx.x * blockDim.x + threadIdx.x;
  if (t < 192 * 64) ipw16[t] = (_Float16)ipw[t];
  if (t < 64 * 64)  opw16[t] = (_Float16)opw[t];
}

// ---------------- QKV projection GEMM: [57600,64] x [192,64]^T -> f16 [57600,192] ----------------
// A tile (16x64 f16, 2KB) staged into LDS via the Tensor Data Mover when available.
__global__ void k_qkv(const _Float16* __restrict__ xh, const _Float16* __restrict__ w16,
                      const float* __restrict__ bias, _Float16* __restrict__ qkvh) {
  __shared__ _Float16 tA[16 * 64];
  int lane = threadIdx.x & 31; int hl = lane >> 4; int r = lane & 15;
  int m0 = blockIdx.x * 16; int n0 = blockIdx.y * 16;

#if __has_builtin(__builtin_amdgcn_tensor_load_to_lds)
  {
    // Tensor DMA descriptor (D#): 2D tensor [57600 rows x 64 cols] of 2-byte elems,
    // tile = 64 x 16 starting at row m0, into LDS offset 0.
    unsigned long long ga = (unsigned long long)(const void*)(xh + (size_t)m0 * 64);
    tdm_v4u g0;
    g0[0] = 1u;                                             // count=1, user mode
    g0[1] = 0u;                                             // lds_addr = 0
    g0[2] = (unsigned)(ga & 0xFFFFFFFFu);                   // global_addr[31:0]
    g0[3] = (unsigned)((ga >> 32) & 0x01FFFFFFu) | (2u << 30); // global_addr[56:32] | type=2
    tdm_v8i g1;
    g1[0] = 0x00010000;            // workgroup_mask=0, data_size=1 (2 bytes)
    g1[1] = (64 << 16);            // tensor_dim0[15:0]=64  (bits 79:48 low half)
    g1[2] = (int)0xE1000000;       // tensor_dim1[15:0]=57600 in bits[31:16]
    g1[3] = (64 << 16);            // tile_dim0=64 (bits 127:112)
    g1[4] = 16;                    // tile_dim1=16
    g1[5] = 64;                    // tensor_dim0_stride=64
    g1[6] = 0; g1[7] = 0;
    tdm_v4i gz4 = {0, 0, 0, 0};
    tdm_v8i gz8 = {0, 0, 0, 0, 0, 0, 0, 0};
    __builtin_amdgcn_tensor_load_to_lds(g0, g1, gz4, gz4, gz8, 0);
    __builtin_amdgcn_s_wait_tensorcnt(0);
    // The TDM write to LDS is invisible to alias analysis (the LDS address is an
    // integer inside the descriptor). Escape tA's address into opaque asm with a
    // memory clobber so the compiler keeps the allocation, treats it as written,
    // and does not hoist the ds_loads above the tensorcnt wait.
    asm volatile("" : : "v"((const void*)&tA[0]) : "memory");
  }
#else
  for (int e = lane; e < 16 * 64; e += 32) tA[e] = xh[(size_t)m0 * 64 + e];
#endif

  v8f c = {};
  for (int kb = 0; kb < 64; kb += 32) {
    v16h a, bf;
    int n = n0 + r;
    #pragma unroll
    for (int j = 0; j < 16; ++j) {
      int ka = kb + ((j >> 3) << 4) + hl * 8 + (j & 7);   // A 16x32 f16 layout
      int kk = kb + ((j >> 3) << 4) + hl * 8 + (j & 7);   // B 32x16 f16 layout
      a[j]  = tA[r * 64 + ka];
      bf[j] = w16[(size_t)n * 64 + kk];                   // Bmat[k][n] = W[n][k]
    }
    c = wmma_f16(a, bf, c);
  }
  #pragma unroll
  for (int vi = 0; vi < 8; ++vi) {
    int m = m0 + vi + 8 * hl; int n = n0 + r;
    qkvh[(size_t)m * 192 + n] = (_Float16)(c[vi] + bias[n]);
  }
}

// ---------------- attention: scores+softmax+ctx+color_weights ----------------
__global__ __launch_bounds__(256) void k_attn(const _Float16* __restrict__ qkvh,
                                              _Float16* __restrict__ ctxh,
                                              float* __restrict__ cw) {
  __shared__ float S[16][SCOL];
  float* Sf = &S[0][0];
  int b = blockIdx.x / QT_, qt = blockIdx.x % QT_;
  int q0 = qt * 16;
  int tid = threadIdx.x; int wv = tid >> 5; int lane = tid & 31;
  int hl = lane >> 4; int r = lane & 15;

  // per-thread head-mean accumulators for color_weights (write global exactly once)
  float cwacc[CWPT];
  #pragma unroll
  for (int i = 0; i < CWPT; ++i) cwacc[i] = 0.f;

  // zero the padded tail columns once (rows 0-1 later overwritten by phase D)
  for (int e = tid; e < 16 * (SCOL - Lq); e += 256) {
    int rr = e / (SCOL - Lq); int cc = Lq + e % (SCOL - Lq);
    S[rr][cc] = 0.f;
  }
  __syncthreads();

  for (int h = 0; h < NH_; ++h) {
    // hint the V slice for this head into cache
    __builtin_prefetch((const void*)(qkvh + ((size_t)b * Lq) * 192 + 128 + h * HD_), 0, 1);

    // ---- scores: S[m][n] = 0.25 * sum_k Q[m,k]*K[n,k], tiles split over waves ----
    for (int t = wv; t < QT_; t += 8) {
      int n0 = t * 16;
      v16h a, bf;
      int m = q0 + r, n = n0 + r;
      #pragma unroll
      for (int j = 0; j < 16; ++j) {
        int ka = ((j >> 3) << 4) + hl * 8 + (j & 7);   // K index for A frag (valid < 16)
        int kk = ((j >> 3) << 4) + hl * 8 + (j & 7);   // K index for B frag
        a[j]  = (m < Lq && ka < HD_)
                  ? qkvh[((size_t)(b * Lq + m)) * 192 + h * HD_ + ka] : (_Float16)0.f;
        bf[j] = (n < Lq && kk < HD_)
                  ? qkvh[((size_t)(b * Lq + n)) * 192 + 64 + h * HD_ + kk] : (_Float16)0.f;
      }
      v8f c = {};
      c = wmma_f16(a, bf, c);
      #pragma unroll
      for (int vi = 0; vi < 8; ++vi) {
        int ml = vi + 8 * hl; int n2 = n0 + r;
        if (n2 < Lq) S[ml][n2] = c[vi] * 0.25f;        // 1/sqrt(HD)
      }
    }
    __syncthreads();

    // ---- softmax per row, 2 rows per wave ----
    for (int rr = wv * 2; rr < wv * 2 + 2; ++rr) {
      int grow = q0 + rr;
      if (grow < Lq) {
        float mx = -3.0e38f;
        for (int cc = lane; cc < Lq; cc += 32) mx = fmaxf(mx, S[rr][cc]);
        for (int off = 16; off > 0; off >>= 1) mx = fmaxf(mx, __shfl_down(mx, off, 32));
        mx = __shfl(mx, 0, 32);
        float sm = 0.f;
        for (int cc = lane; cc < Lq; cc += 32) {
          float e = __expf(S[rr][cc] - mx); S[rr][cc] = e; sm += e;
        }
        for (int off = 16; off > 0; off >>= 1) sm += __shfl_down(sm, off, 32);
        sm = __shfl(sm, 0, 32);
        float inv = 1.f / sm;
        for (int cc = lane; cc < Lq; cc += 32) S[rr][cc] *= inv;
      }
    }
    __syncthreads();

    // ---- phase C: cw accumulate in registers + per-wave ctx partials ----
    #pragma unroll
    for (int i = 0; i < CWPT; ++i) {
      int e = tid + i * 256;
      if (e < 16 * Lq) cwacc[i] += 0.25f * Sf[(e / Lq) * SCOL + (e % Lq)];
    }
    v8f pacc = {};
    for (int kb = wv * 32; kb < SCOL; kb += 256) {   // K chunks strided over 8 waves
      v16h a, bf;
      #pragma unroll
      for (int j = 0; j < 16; ++j) {
        int ka = kb + ((j >> 3) << 4) + hl * 8 + (j & 7);
        a[j] = (_Float16)S[r][ka];                   // pad cols: 0, or stale in rows 0-1; B is 0 there
        int key = kb + ((j >> 3) << 4) + hl * 8 + (j & 7);
        bf[j] = (key < Lq)
                  ? qkvh[((size_t)(b * Lq + key)) * 192 + 128 + h * HD_ + r] : (_Float16)0.f;
      }
      pacc = wmma_f16(a, bf, pacc);
    }
    __syncthreads();

    // ---- phase D: dump ctx partials into (reused) score LDS ----
    #pragma unroll
    for (int vi = 0; vi < 8; ++vi) Sf[wv * 256 + vi * 32 + lane] = pacc[vi];
    __syncthreads();

    // ---- phase E: wave 0 reduces partials and stores ctx ----
    if (wv == 0) {
      #pragma unroll
      for (int vi = 0; vi < 8; ++vi) {
        float s = 0.f;
        #pragma unroll
        for (int w = 0; w < 8; ++w) s += Sf[w * 256 + vi * 32 + lane];
        int grow = q0 + vi + 8 * hl;
        if (grow < Lq)
          ctxh[((size_t)(b * Lq + grow)) * Dm + h * HD_ + r] = (_Float16)s;
      }
    }
    __syncthreads();
  }

  // ---- single streaming store of mean attention weights ----
  #pragma unroll
  for (int i = 0; i < CWPT; ++i) {
    int e = tid + i * 256;
    if (e < 16 * Lq) {
      int rr = e / Lq, cc = e % Lq;
      int grow = q0 + rr;
      if (grow < Lq) cw[((size_t)b * Lq + grow) * Lq + cc] = cwacc[i];
    }
  }
}

// ---------------- out-projection GEMM: [57600,64] x [64,64]^T -> f32 attended ----------------
__global__ void k_outproj(const _Float16* __restrict__ ctxh, const _Float16* __restrict__ w16,
                          const float* __restrict__ bias, float* __restrict__ attended) {
  int lane = threadIdx.x & 31; int hl = lane >> 4; int r = lane & 15;
  int m0 = blockIdx.x * 16; int n0 = blockIdx.y * 16;
  v8f c = {};
  for (int kb = 0; kb < 64; kb += 32) {
    v16h a, bf;
    int m = m0 + r; int n = n0 + r;
    #pragma unroll
    for (int j = 0; j < 16; ++j) {
      int ka = kb + ((j >> 3) << 4) + hl * 8 + (j & 7);
      int kk = kb + ((j >> 3) << 4) + hl * 8 + (j & 7);
      a[j]  = ctxh[(size_t)m * 64 + ka];
      bf[j] = w16[(size_t)n * 64 + kk];
    }
    c = wmma_f16(a, bf, c);
  }
  #pragma unroll
  for (int vi = 0; vi < 8; ++vi) {
    int m = m0 + vi + 8 * hl; int n = n0 + r;
    attended[(size_t)m * 64 + n] = c[vi] + bias[n];
  }
}

// ---------------- conv1: 10->32, 3x3 SAME, relu ----------------
__global__ void k_conv1(const float* __restrict__ in, const float* __restrict__ w,
                        const float* __restrict__ bias, float* __restrict__ out) {
  int t = blockIdx.x * blockDim.x + threadIdx.x;
  if (t >= B_ * 32 * HW_) return;
  int s = t % HW_; int tmp = t / HW_; int co = tmp % 32; int b = tmp / 32;
  int y = s / 30, x = s % 30;
  float acc = bias[co];
  for (int ci = 0; ci < 10; ++ci) {
    #pragma unroll
    for (int ky = 0; ky < 3; ++ky) {
      int yy = y + ky - 1; if (yy < 0 || yy >= 30) continue;
      #pragma unroll
      for (int kx = 0; kx < 3; ++kx) {
        int xx = x + kx - 1; if (xx < 0 || xx >= 30) continue;
        acc += in[((size_t)(b * 10 + ci)) * HW_ + yy * 30 + xx]
             * w[(((co * 10) + ci) * 3 + ky) * 3 + kx];
      }
    }
  }
  out[t] = fmaxf(acc, 0.f);
}

// ---------------- conv2: 32->64, 3x3 SAME, relu, + attended (combined) ----------------
__global__ void k_conv2(const float* __restrict__ s1, const float* __restrict__ w,
                        const float* __restrict__ bias, const float* __restrict__ attended,
                        float* __restrict__ combined) {
  int t = blockIdx.x * blockDim.x + threadIdx.x;
  if (t >= B_ * 64 * HW_) return;
  int s = t % HW_; int tmp = t / HW_; int co = tmp % 64; int b = tmp / 64;
  int y = s / 30, x = s % 30;
  float acc = bias[co];
  for (int ci = 0; ci < 32; ++ci) {
    #pragma unroll
    for (int ky = 0; ky < 3; ++ky) {
      int yy = y + ky - 1; if (yy < 0 || yy >= 30) continue;
      #pragma unroll
      for (int kx = 0; kx < 3; ++kx) {
        int xx = x + kx - 1; if (xx < 0 || xx >= 30) continue;
        acc += s1[((size_t)(b * 32 + ci)) * HW_ + yy * 30 + xx]
             * w[(((co * 32) + ci) * 3 + ky) * 3 + kx];
      }
    }
  }
  combined[t] = fmaxf(acc, 0.f) + attended[((size_t)(b * HW_ + s)) * 64 + co];
}

// ---------------- global mean pool ----------------
__global__ void k_pool(const float* __restrict__ combined, float* __restrict__ gf) {
  int t = blockIdx.x * blockDim.x + threadIdx.x;
  if (t >= B_ * 64) return;
  const float* p = combined + (size_t)t * HW_;
  float s = 0.f;
  for (int i = 0; i < HW_; ++i) s += p[i];
  gf[t] = s * (1.f / 900.f);
}

// ---------------- MLP + softmax color map + argmax new_color ----------------
__global__ __launch_bounds__(128) void k_mlp(const float* __restrict__ gf,
                                             const float* __restrict__ l1w, const float* __restrict__ l1b,
                                             const float* __restrict__ l2w, const float* __restrict__ l2b,
                                             float* __restrict__ cmap, int* __restrict__ newc) {
  __shared__ float g[64]; __shared__ float hm[128]; __shared__ float lg[100];
  int b = blockIdx.x; int t = threadIdx.x;
  if (t < 64) g[t] = gf[b * 64 + t];
  __syncthreads();
  {
    float a = l1b[t];
    for (int k = 0; k < 64; ++k) a += g[k] * l1w[t * 64 + k];
    hm[t] = fmaxf(a, 0.f);
  }
  __syncthreads();
  if (t < 100) {
    float a = l2b[t];
    for (int k = 0; k < 128; ++k) a += hm[k] * l2w[t * 128 + k];
    lg[t] = a;
  }
  __syncthreads();
  if (t < 10) {
    float mx = lg[t * 10]; int bi = 0;
    #pragma unroll
    for (int j = 1; j < 10; ++j) { float v = lg[t * 10 + j]; if (v > mx) { mx = v; bi = j; } }
    float e[10]; float sm = 0.f;
    #pragma unroll
    for (int j = 0; j < 10; ++j) { e[j] = __expf(lg[t * 10 + j] - mx); sm += e[j]; }
    float inv = 1.f / sm;
    #pragma unroll
    for (int j = 0; j < 10; ++j) cmap[(size_t)b * 100 + t * 10 + j] = e[j] * inv;
    newc[b * 10 + t] = bi;
  }
}

// ---------------- one-hot color remap ----------------
__global__ void k_map(const int* __restrict__ idx, const int* __restrict__ newc,
                      float* __restrict__ mo) {
  int t = blockIdx.x * blockDim.x + threadIdx.x;
  if (t >= B_ * HW_) return;
  int s = t % HW_, b = t / HW_;
  int nc = newc[b * 10 + idx[t]];
  #pragma unroll
  for (int c = 0; c < 10; ++c)
    mo[((size_t)(b * 10 + c)) * HW_ + s] = (c == nc) ? 1.f : 0.f;
}

// ---------------- decoder convT(64->32) as conv w/ flipped kernel + BN + relu ----------------
__global__ void k_dec1(const float* __restrict__ combined, const float* __restrict__ w,
                       const float* __restrict__ bias, const float* __restrict__ bng,
                       const float* __restrict__ bnb, float* __restrict__ dbuf) {
  int t = blockIdx.x * blockDim.x + threadIdx.x;
  if (t >= B_ * 32 * HW_) return;
  int s = t % HW_; int tmp = t / HW_; int o = tmp % 32; int b = tmp / 32;
  int y = s / 30, x = s % 30;
  float acc = bias[o];
  for (int ci = 0; ci < 64; ++ci) {
    #pragma unroll
    for (int ky = 0; ky < 3; ++ky) {
      int yy = y + ky - 1; if (yy < 0 || yy >= 30) continue;
      #pragma unroll
      for (int kx = 0; kx < 3; ++kx) {
        int xx = x + kx - 1; if (xx < 0 || xx >= 30) continue;
        // w1[o][ci][ky][kx] = dec1_w[ci][o][2-ky][2-kx]
        acc += combined[((size_t)(b * 64 + ci)) * HW_ + yy * 30 + xx]
             * w[(((ci * 32) + o) * 3 + (2 - ky)) * 3 + (2 - kx)];
      }
    }
  }
  const float inv = 0.99999500003749968752f; // 1/sqrt(1+1e-5)
  float d = acc * inv * bng[o] + bnb[o];
  dbuf[t] = fmaxf(d, 0.f);
}

// ---------------- predicted = einsum(d, dec2_w) + dec2_b ----------------
__global__ void k_pred(const float* __restrict__ dbuf, const float* __restrict__ w2,
                       const float* __restrict__ b2, float* __restrict__ out) {
  int t = blockIdx.x * blockDim.x + threadIdx.x;
  if (t >= B_ * 10 * HW_) return;
  int s = t % HW_; int tmp = t / HW_; int o = tmp % 10; int b = tmp / 10;
  float acc = b2[o];
  #pragma unroll
  for (int i = 0; i < 32; ++i)
    acc += dbuf[((size_t)(b * 32 + i)) * HW_ + s] * w2[i * 10 + o];
  out[((size_t)(b * 10 + o)) * HW_ + s] = acc;
}

static inline size_t al256(size_t x) { return (x + 255) & ~(size_t)255; }

extern "C" void kernel_launch(void* const* d_in, const int* in_sizes, int n_in,
                              void* d_out, int out_size, void* d_ws, size_t ws_size,
                              hipStream_t stream) {
  const float* input_grid = (const float*)d_in[0];
  const float* embed      = (const float*)d_in[1];
  const float* in_proj_w  = (const float*)d_in[2];
  const float* in_proj_b  = (const float*)d_in[3];
  const float* out_proj_w = (const float*)d_in[4];
  const float* out_proj_b = (const float*)d_in[5];
  const float* conv1_w    = (const float*)d_in[6];
  const float* conv1_b    = (const float*)d_in[7];
  const float* conv2_w    = (const float*)d_in[8];
  const float* conv2_b    = (const float*)d_in[9];
  const float* lin1_w     = (const float*)d_in[10];
  const float* lin1_b     = (const float*)d_in[11];
  const float* lin2_w     = (const float*)d_in[12];
  const float* lin2_b     = (const float*)d_in[13];
  const float* dec1_w     = (const float*)d_in[14];
  const float* dec1_b     = (const float*)d_in[15];
  const float* bn_g       = (const float*)d_in[16];
  const float* bn_b       = (const float*)d_in[17];
  const float* dec2_w     = (const float*)d_in[18];
  const float* dec2_b     = (const float*)d_in[19];

  float* out  = (float*)d_out;
  float* pred = out + OUT_PRED;
  float* cmap = out + OUT_CMAP;
  float* cw   = out + OUT_CW;
  float* mo   = out + OUT_MAP;

  // workspace carve-out
  char* ws = (char*)d_ws;
  size_t o = 0;
  int*       idx      = (int*)      (ws + o); o = al256(o + (size_t)B_ * HW_ * 4);
  int*       newc     = (int*)      (ws + o); o = al256(o + (size_t)B_ * 10 * 4);
  _Float16*  xh       = (_Float16*) (ws + o); o = al256(o + (size_t)B_ * HW_ * Dm * 2);
  _Float16*  ipw16    = (_Float16*) (ws + o); o = al256(o + (size_t)192 * 64 * 2);
  _Float16*  opw16    = (_Float16*) (ws + o); o = al256(o + (size_t)64 * 64 * 2);
  _Float16*  qkvh     = (_Float16*) (ws + o); o = al256(o + (size_t)B_ * HW_ * 192 * 2);
  _Float16*  ctxh     = (_Float16*) (ws + o); o = al256(o + (size_t)B_ * HW_ * Dm * 2);
  float*     attended = (float*)    (ws + o); o = al256(o + (size_t)B_ * HW_ * Dm * 4);
  float*     s1       = (float*)    (ws + o); o = al256(o + (size_t)B_ * 32 * HW_ * 4);
  float*     combined = (float*)    (ws + o); o = al256(o + (size_t)B_ * 64 * HW_ * 4);
  float*     gf       = (float*)    (ws + o); o = al256(o + (size_t)B_ * 64 * 4);
  float*     dbuf     = (float*)    (ws + o); o = al256(o + (size_t)B_ * 32 * HW_ * 4);
  (void)ws_size; (void)out_size; (void)n_in; (void)in_sizes;

  const int T = 256;
  k_argmax<<<(B_ * HW_ + T - 1) / T, T, 0, stream>>>(input_grid, idx);
  k_gather<<<(B_ * HW_ * Dm + T - 1) / T, T, 0, stream>>>(embed, idx, xh);
  k_cvt<<<(192 * 64 + T - 1) / T, T, 0, stream>>>(in_proj_w, out_proj_w, ipw16, opw16);

  k_qkv<<<dim3(B_ * HW_ / 16, 192 / 16), 32, 0, stream>>>(xh, ipw16, in_proj_b, qkvh);
  k_attn<<<B_ * QT_, 256, 0, stream>>>(qkvh, ctxh, cw);
  k_outproj<<<dim3(B_ * HW_ / 16, 64 / 16), 32, 0, stream>>>(ctxh, opw16, out_proj_b, attended);

  k_conv1<<<(B_ * 32 * HW_ + T - 1) / T, T, 0, stream>>>(input_grid, conv1_w, conv1_b, s1);
  k_conv2<<<(B_ * 64 * HW_ + T - 1) / T, T, 0, stream>>>(s1, conv2_w, conv2_b, attended, combined);
  k_pool<<<(B_ * 64 + T - 1) / T, T, 0, stream>>>(combined, gf);
  k_mlp<<<B_, 128, 0, stream>>>(gf, lin1_w, lin1_b, lin2_w, lin2_b, cmap, newc);
  k_map<<<(B_ * HW_ + T - 1) / T, T, 0, stream>>>(idx, newc, mo);
  k_dec1<<<(B_ * 32 * HW_ + T - 1) / T, T, 0, stream>>>(combined, dec1_w, dec1_b, bn_g, bn_b, dbuf);
  k_pred<<<(B_ * 10 * HW_ + T - 1) / T, T, 0, stream>>>(dbuf, dec2_w, dec2_b, pred);
}
